// TSLSTM_20401094656695
// MI455X (gfx1250) — compile-verified, matrix operand order checked by reading, not run on hardware
//
#include <hip/hip_runtime.h>
#include <hip/hip_bf16.h>
#include <stdint.h>

// ---------------------------------------------------------------------------
// Types for CDNA5 WMMA (wave32, 16x16x32 bf16 -> f32)
// ---------------------------------------------------------------------------
typedef __bf16 bf16_t;
typedef __attribute__((ext_vector_type(16))) __bf16 v16bf;
typedef __attribute__((ext_vector_type(8)))  float  v8f;
typedef __attribute__((ext_vector_type(4)))  uint32_t u32x4;
typedef __attribute__((ext_vector_type(4)))  int32_t  i32x4;
typedef __attribute__((ext_vector_type(8)))  int32_t  i32x8;

union FragU { v16bf v; float4 q[2]; };

#if defined(__gfx1250__) && __has_builtin(__builtin_amdgcn_tensor_load_to_lds)
#define USE_TDM 1
#else
#define USE_TDM 0
#endif

// LDS tile row stride: 32 data bf16 + 8 pad bf16 = 80 bytes (16B aligned,
// 20-bank shift per row -> conflict-free b128 fragment reads).
#define LROW 40

__device__ __forceinline__ v8f wmma_bf16(v16bf a, v16bf b, v8f c) {
    return __builtin_amdgcn_wmma_f32_16x16x32_bf16(
        false, a, false, b, (short)0, c, false, false);
}

// Fragment from a padded LDS tile: lane L holds row (L&15); lanes 0-15 hold
// k in [0,8)+[16,24), lanes 16-31 hold k in [8,16)+[24,32)  (ISA 16-bit A/B).
__device__ __forceinline__ v16bf lds_frag(const bf16_t* p, int lane) {
    const bf16_t* q = p + (size_t)(lane & 15) * LROW + ((lane >> 4) << 3);
    FragU u;
    u.q[0] = *(const float4*)(q);
    u.q[1] = *(const float4*)(q + 16);
    return u.v;
}

// Fragment straight from a row-major global matrix (recurrent step kernel).
__device__ __forceinline__ v16bf load_frag(const bf16_t* __restrict__ base,
                                           size_t ld, int lane) {
    const bf16_t* p = base + (size_t)(lane & 15) * ld + ((lane >> 4) << 3);
    FragU u;
    u.q[0] = *(const float4*)(p);
    u.q[1] = *(const float4*)(p + 16);
    return u.v;
}

#if USE_TDM
// ---------------------------------------------------------------------------
// Tensor Data Mover: stage one 2D bf16 tile (rows x 32 elems, arbitrary row
// stride) global -> LDS with 16B row padding (pad_interval=16dw, amount=4dw).
// D# layout per CDNA5 ISA (type=2, data_size -> 2B).
// ---------------------------------------------------------------------------
__device__ __forceinline__ void tdm_load_2d(uint32_t lds_off,
                                            const bf16_t* gptr,
                                            uint32_t rows,
                                            uint64_t row_stride_elems,
                                            uint32_t tensor_d0,
                                            uint32_t tensor_d1) {
    uint64_t ga = (uint64_t)(uintptr_t)gptr;
    u32x4 g0;
    g0[0] = 1u;                                        // count=1 (valid), user
    g0[1] = lds_off;                                   // lds_addr
    g0[2] = (uint32_t)(ga & 0xFFFFFFFFu);              // global_addr[31:0]
    g0[3] = (uint32_t)((ga >> 32) & 0x1FFFFFFu) | (2u << 30);  // [56:32]|type=2
    i32x8 g1;
    g1[0] = (int)((1u << 16)      // data_size = 2 bytes
                | (1u << 20)      // pad_enable
                | (3u << 22)      // pad_interval: 16 dwords (64B)
                | (3u << 25));    // pad_amount:   4 dwords (16B)
    g1[1] = (int)((tensor_d0 & 0xFFFFu) << 16);                  // dim0[15:0]
    g1[2] = (int)(((tensor_d0 >> 16) & 0xFFFFu)
               |  ((tensor_d1 & 0xFFFFu) << 16));                // dim0 hi|dim1 lo
    g1[3] = (int)(((tensor_d1 >> 16) & 0xFFFFu) | (32u << 16));  // dim1 hi|tile_dim0=32
    g1[4] = (int)(rows & 0xFFFFu);                               // tile_dim1, tile_dim2=0
    g1[5] = (int)(uint32_t)(row_stride_elems & 0xFFFFFFFFu);     // dim0_stride lo
    g1[6] = (int)(uint32_t)((row_stride_elems >> 32) & 0xFFFFu); // dim0_stride hi
    g1[7] = 0;
    i32x4 z4 = {0, 0, 0, 0};
#if defined(__clang_major__) && __clang_major__ >= 23
    i32x8 z8 = {0, 0, 0, 0, 0, 0, 0, 0};
    __builtin_amdgcn_tensor_load_to_lds(g0, g1, z4, z4, z8, 0);
#else
    __builtin_amdgcn_tensor_load_to_lds(g0, g1, z4, z4, 0);
#endif
}
#endif  // USE_TDM

// ---------------------------------------------------------------------------
// fp32 -> bf16 weight conversion
// ---------------------------------------------------------------------------
__global__ void f32_to_bf16_kernel(const float* __restrict__ in,
                                   bf16_t* __restrict__ out, size_t n) {
    size_t i = (size_t)blockIdx.x * blockDim.x + threadIdx.x;
    if (i < n) out[i] = (bf16_t)in[i];
}

// ---------------------------------------------------------------------------
// Embedding -> binary spikes (renorm scale > 0 preserves sign; row 0 -> 0).
// ---------------------------------------------------------------------------
__global__ void embed_spike_kernel(const int* __restrict__ data,
                                   const float* __restrict__ emb,
                                   bf16_t* __restrict__ x, int D) {
    int tb = blockIdx.x;
    int row = data[tb];
    const float* e = emb + (size_t)row * D;
    bool nz = (row != 0);
    for (int d = threadIdx.x; d < D; d += blockDim.x) {
        float v = e[d];
        x[(size_t)tb * D + d] = (nz && v > 0.f) ? (bf16_t)1.f : (bf16_t)0.f;
    }
}

// ---------------------------------------------------------------------------
// WMMA GEMM, TDM-staged double-buffered LDS tiles:
//   C[M,N] = A[M,K] @ W[N,K]^T + bias0 + bias1
// Block = 128 threads (4 waves), block tile 128Mx64N, K-step 32.
// Wave w owns M-subtiles {2w, 2w+1}; 4 B-fragments shared by 8 WMMAs.
// ---------------------------------------------------------------------------
template<bool OUT_BF16>
__global__ void gemm_lds_kernel(const bf16_t* __restrict__ A,
                                const bf16_t* __restrict__ W,
                                const float* __restrict__ bias0,
                                const float* __restrict__ bias1,
                                void* __restrict__ Cout,
                                int M, int N, int K) {
    __shared__ __align__(16) bf16_t Abuf[2][128][LROW];
    __shared__ __align__(16) bf16_t Bbuf[2][64][LROW];
    const int lane = threadIdx.x & 31;
    const int wave = threadIdx.x >> 5;
    const int mb = blockIdx.x * 128;
    const int nb = blockIdx.y * 64;
    const int nk = K >> 5;
    v8f acc[2][4] = {};

#if USE_TDM
    if (threadIdx.x < 32) {
        tdm_load_2d((uint32_t)(uintptr_t)&Abuf[0][0][0], A + (size_t)mb * K,
                    128, (uint64_t)K, (uint32_t)K, 128);
        tdm_load_2d((uint32_t)(uintptr_t)&Bbuf[0][0][0], W + (size_t)nb * K,
                    64, (uint64_t)K, (uint32_t)K, 64);
    }
    for (int kt = 0; kt < nk; ++kt) {
        const int cur = kt & 1;
        if (threadIdx.x < 32) {
            if (kt + 1 < nk) {
                int k0 = (kt + 1) << 5;
                tdm_load_2d((uint32_t)(uintptr_t)&Abuf[cur ^ 1][0][0],
                            A + (size_t)mb * K + k0, 128, (uint64_t)K,
                            (uint32_t)(K - k0), 128);
                tdm_load_2d((uint32_t)(uintptr_t)&Bbuf[cur ^ 1][0][0],
                            W + (size_t)nb * K + k0, 64, (uint64_t)K,
                            (uint32_t)(K - k0), 64);
                __builtin_amdgcn_s_wait_tensorcnt(2);  // current buffers ready
            } else {
                __builtin_amdgcn_s_wait_tensorcnt(0);
            }
        }
        __syncthreads();
        v16bf a0 = lds_frag(&Abuf[cur][wave * 32][0], lane);
        v16bf a1 = lds_frag(&Abuf[cur][wave * 32 + 16][0], lane);
#pragma unroll
        for (int j = 0; j < 4; ++j) {
            v16bf b = lds_frag(&Bbuf[cur][j * 16][0], lane);
            acc[0][j] = wmma_bf16(a0, b, acc[0][j]);
            acc[1][j] = wmma_bf16(a1, b, acc[1][j]);
        }
        __syncthreads();  // tile consumed before wave0 re-targets it
    }
#else
    for (int kt = 0; kt < nk; ++kt) {
        int k0 = kt << 5;
        for (int i = threadIdx.x; i < 512; i += 128) {
            int row = i >> 2, ch = (i & 3) * 8;
            *(float4*)&Abuf[0][row][ch] = *(const float4*)(A + (size_t)(mb + row) * K + k0 + ch);
            if (row < 64)
                *(float4*)&Bbuf[0][row][ch] = *(const float4*)(W + (size_t)(nb + row) * K + k0 + ch);
        }
        __syncthreads();
        v16bf a0 = lds_frag(&Abuf[0][wave * 32][0], lane);
        v16bf a1 = lds_frag(&Abuf[0][wave * 32 + 16][0], lane);
#pragma unroll
        for (int j = 0; j < 4; ++j) {
            v16bf b = lds_frag(&Bbuf[0][j * 16][0], lane);
            acc[0][j] = wmma_bf16(a0, b, acc[0][j]);
            acc[1][j] = wmma_bf16(a1, b, acc[1][j]);
        }
        __syncthreads();
    }
#endif

#pragma unroll
    for (int s = 0; s < 2; ++s) {
        const int m0 = mb + wave * 32 + s * 16;
#pragma unroll
        for (int j = 0; j < 4; ++j) {
            int col = nb + 16 * j + (lane & 15);
            float bv = (bias0 ? bias0[col] : 0.f) + (bias1 ? bias1[col] : 0.f);
#pragma unroll
            for (int r = 0; r < 8; ++r) {
                int row = m0 + r + ((lane >> 4) << 3);
                float v = acc[s][j][r] + bv;
                if (OUT_BF16) ((bf16_t*)Cout)[(size_t)row * N + col] = (bf16_t)v;
                else          ((float*)Cout)[(size_t)row * N + col]  = v;
            }
        }
    }
}

// ---------------------------------------------------------------------------
// One sLSTM recurrence step (B=32 rows, D columns tiled by 16):
//   gates = xg[t] + mem @ whh^T ; i,f,g,o split along 4D
// Block = 64 threads (2 waves), grid = D/16; mem double-buffered in bf16.
// Fragment loads software-pipelined one k-step ahead of the WMMAs.
// ---------------------------------------------------------------------------
__global__ void slstm_step_kernel(const float* __restrict__ xg_t,
                                  const bf16_t* __restrict__ whh,
                                  const bf16_t* __restrict__ mem_in,
                                  bf16_t* __restrict__ mem_out,
                                  float* __restrict__ mem_f,
                                  float* __restrict__ syn_f,
                                  bf16_t* __restrict__ spk_out,
                                  const float* __restrict__ thr_p,
                                  int D) {
    const int lane = threadIdx.x & 31;
    const int wave = threadIdx.x >> 5;
    const int b0 = wave * 16;
    const int d0 = blockIdx.x * 16;
    const int D4 = 4 * D;
    const float thr = *thr_p;

    v8f acc[4] = {};
    v16bf a = load_frag(mem_in + (size_t)b0 * D, D, lane);
    v16bf bw[4];
#pragma unroll
    for (int g = 0; g < 4; ++g)
        bw[g] = load_frag(whh + (size_t)(g * D + d0) * D, D, lane);
    for (int k0 = 32; k0 < D; k0 += 32) {
        v16bf an = load_frag(mem_in + (size_t)b0 * D + k0, D, lane);
        v16bf bn[4];
#pragma unroll
        for (int g = 0; g < 4; ++g)
            bn[g] = load_frag(whh + (size_t)(g * D + d0) * D + k0, D, lane);
#pragma unroll
        for (int g = 0; g < 4; ++g)
            acc[g] = wmma_bf16(a, bw[g], acc[g]);
        a = an;
#pragma unroll
        for (int g = 0; g < 4; ++g) bw[g] = bn[g];
    }
#pragma unroll
    for (int g = 0; g < 4; ++g)
        acc[g] = wmma_bf16(a, bw[g], acc[g]);

    const int d = d0 + (lane & 15);
#pragma unroll
    for (int r = 0; r < 8; ++r) {
        int b = b0 + r + ((lane >> 4) << 3);
        size_t idx = (size_t)b * D + d;
        size_t xb  = (size_t)b * D4 + d;
        float xi = acc[0][r] + xg_t[xb + 0 * D];
        float xf = acc[1][r] + xg_t[xb + 1 * D];
        float xg = acc[2][r] + xg_t[xb + 2 * D];
        float xo = acc[3][r] + xg_t[xb + 3 * D];
        float synv = syn_f[idx];
        float memv = mem_f[idx];
        float reset = (memv > thr) ? 1.f : 0.f;
        float si = 1.f / (1.f + __expf(-xi));
        float sf = 1.f / (1.f + __expf(-xf));
        float so = 1.f / (1.f + __expf(-xo));
        synv = sf * synv + si * tanhf(xg);
        memv = so * tanhf(synv) - reset * thr;
        syn_f[idx] = synv;
        mem_f[idx] = memv;
        mem_out[idx] = (bf16_t)memv;
        spk_out[idx] = (memv - thr > 0.f) ? (bf16_t)1.f : (bf16_t)0.f;
    }
}

// ---------------------------------------------------------------------------
// Decoder: Z[t,v] = spk[t,b,:] @ dec_w[v,:]^T + dec_b[v], fused leaky-IF scan.
// Block = (2 batches, 64-v strip). Wave w owns (b_local = w&1, t-tiles
// {2*(w>>1), 2*(w>>1)+1}); B-fragments shared by 8 WMMAs per wave.
// Three strided TDM descriptors per k-step (A for each b, plus B).
// ---------------------------------------------------------------------------
__global__ void decoder_kernel(const bf16_t* __restrict__ spk,   // T x B x D
                               const bf16_t* __restrict__ dec_w, // V x D
                               const float* __restrict__ dec_b,  // V
                               float* __restrict__ out,          // T x B x V
                               const float* __restrict__ beta_p,
                               int B, int D, int V, int T) {
    __shared__ __align__(16) bf16_t Abuf[2][2][64][LROW];
    __shared__ __align__(16) bf16_t Bbuf[2][64][LROW];
    __shared__ float zs[2][64][65];
    const int lane = threadIdx.x & 31;
    const int wave = threadIdx.x >> 5;
    const int bl   = wave & 1;          // which batch this wave computes
    const int tp   = (wave >> 1) * 2;   // first of two t-subtiles
    const int b0 = blockIdx.x * 2;
    const int v0 = blockIdx.y * 64;
    const size_t ldA = (size_t)B * D;   // stride between timesteps
    const int nk = D >> 5;
    v8f acc[2][4] = {};

#if USE_TDM
    if (threadIdx.x < 32) {
        tdm_load_2d((uint32_t)(uintptr_t)&Abuf[0][0][0][0],
                    spk + (size_t)b0 * D, 64, ldA, (uint32_t)D, 64);
        tdm_load_2d((uint32_t)(uintptr_t)&Abuf[0][1][0][0],
                    spk + (size_t)(b0 + 1) * D, 64, ldA, (uint32_t)D, 64);
        tdm_load_2d((uint32_t)(uintptr_t)&Bbuf[0][0][0],
                    dec_w + (size_t)v0 * D, 64, (uint64_t)D, (uint32_t)D, 64);
    }
    for (int kt = 0; kt < nk; ++kt) {
        const int cur = kt & 1;
        if (threadIdx.x < 32) {
            if (kt + 1 < nk) {
                int k0 = (kt + 1) << 5;
                tdm_load_2d((uint32_t)(uintptr_t)&Abuf[cur ^ 1][0][0][0],
                            spk + (size_t)b0 * D + k0, 64, ldA,
                            (uint32_t)(D - k0), 64);
                tdm_load_2d((uint32_t)(uintptr_t)&Abuf[cur ^ 1][1][0][0],
                            spk + (size_t)(b0 + 1) * D + k0, 64, ldA,
                            (uint32_t)(D - k0), 64);
                tdm_load_2d((uint32_t)(uintptr_t)&Bbuf[cur ^ 1][0][0],
                            dec_w + (size_t)v0 * D + k0, 64, (uint64_t)D,
                            (uint32_t)(D - k0), 64);
                __builtin_amdgcn_s_wait_tensorcnt(3);
            } else {
                __builtin_amdgcn_s_wait_tensorcnt(0);
            }
        }
        __syncthreads();
        v16bf a0 = lds_frag(&Abuf[cur][bl][tp * 16][0], lane);
        v16bf a1 = lds_frag(&Abuf[cur][bl][tp * 16 + 16][0], lane);
#pragma unroll
        for (int j = 0; j < 4; ++j) {
            v16bf w = lds_frag(&Bbuf[cur][j * 16][0], lane);
            acc[0][j] = wmma_bf16(a0, w, acc[0][j]);
            acc[1][j] = wmma_bf16(a1, w, acc[1][j]);
        }
        __syncthreads();
    }
#else
    for (int kt = 0; kt < nk; ++kt) {
        int k0 = kt << 5;
        for (int i = threadIdx.x; i < 512; i += 128) {
            int row = i >> 2, ch = (i & 3) * 8;
            int bb = row >> 6, tr = row & 63;
            *(float4*)&Abuf[0][bb][tr][ch] =
                *(const float4*)(spk + (size_t)tr * ldA + (size_t)(b0 + bb) * D + k0 + ch);
            if (row < 64)
                *(float4*)&Bbuf[0][row][ch] = *(const float4*)(dec_w + (size_t)(v0 + row) * D + k0 + ch);
        }
        __syncthreads();
        v16bf a0 = lds_frag(&Abuf[0][bl][tp * 16][0], lane);
        v16bf a1 = lds_frag(&Abuf[0][bl][tp * 16 + 16][0], lane);
#pragma unroll
        for (int j = 0; j < 4; ++j) {
            v16bf w = lds_frag(&Bbuf[0][j * 16][0], lane);
            acc[0][j] = wmma_bf16(a0, w, acc[0][j]);
            acc[1][j] = wmma_bf16(a1, w, acc[1][j]);
        }
        __syncthreads();
    }
#endif

#pragma unroll
    for (int s = 0; s < 2; ++s) {
        const int t0 = (tp + s) * 16;
#pragma unroll
        for (int j = 0; j < 4; ++j) {
            int col = 16 * j + (lane & 15);
            float bv = dec_b[v0 + col];
#pragma unroll
            for (int r = 0; r < 8; ++r) {
                int t = t0 + r + ((lane >> 4) << 3);
                zs[bl][t][col] = acc[s][j][r] + bv;
            }
        }
    }
    __syncthreads();
    {
        const float beta = *beta_p;
        int blc = threadIdx.x >> 6;          // batch within block
        int vv  = threadIdx.x & 63;
        int v = v0 + vv;
        float mem = 0.f;
        for (int t = 0; t < T; ++t) {
            float reset = (mem > 1.0f) ? 1.f : 0.f;
            mem = beta * mem + zs[blc][t][vv] - reset;  // LEAKY_THR = 1
            out[((size_t)t * B + (b0 + blc)) * V + v] = (mem > 1.0f) ? 1.f : 0.f;
        }
    }
}

// ---------------------------------------------------------------------------
// Host orchestration
// ---------------------------------------------------------------------------
static constexpr int T = 64, Bb = 32, Dd = 1024, Vv = 32000;

static inline size_t align256(size_t x) { return (x + 255) & ~(size_t)255; }

extern "C" void kernel_launch(void* const* d_in, const int* in_sizes, int n_in,
                              void* d_out, int out_size, void* d_ws, size_t ws_size,
                              hipStream_t stream) {
    (void)in_sizes; (void)n_in; (void)out_size; (void)ws_size;
    const int*   data  = (const int*)d_in[0];
    const float* emb   = (const float*)d_in[1];
    const float* dec_w = (const float*)d_in[2];
    const float* dec_b = (const float*)d_in[3];
    const float* beta  = (const float*)d_in[4];
    const float* lin_w[2] = { (const float*)d_in[5],  (const float*)d_in[12] };
    const float* lin_b[2] = { (const float*)d_in[6],  (const float*)d_in[13] };
    const float* wih[2]   = { (const float*)d_in[7],  (const float*)d_in[14] };
    const float* whh[2]   = { (const float*)d_in[8],  (const float*)d_in[15] };
    const float* bih[2]   = { (const float*)d_in[9],  (const float*)d_in[16] };
    const float* bhh[2]   = { (const float*)d_in[10], (const float*)d_in[17] };
    const float* thr[2]   = { (const float*)d_in[11], (const float*)d_in[18] };
    float* out = (float*)d_out;

    char* ws = (char*)d_ws;
    size_t off = 0;
    auto take = [&](size_t bytes) { char* p = ws + off; off += align256(bytes); return p; };
    bf16_t* lin_bf[2], *wih_bf[2], *whh_bf[2];
    for (int l = 0; l < 2; ++l) lin_bf[l] = (bf16_t*)take((size_t)Dd * Dd * 2);
    for (int l = 0; l < 2; ++l) wih_bf[l] = (bf16_t*)take((size_t)4 * Dd * Dd * 2);
    for (int l = 0; l < 2; ++l) whh_bf[l] = (bf16_t*)take((size_t)4 * Dd * Dd * 2);
    bf16_t* decw_bf = (bf16_t*)take((size_t)Vv * Dd * 2);
    bf16_t* x_spk   = (bf16_t*)take((size_t)T * Bb * Dd * 2);
    bf16_t* y_bf    = (bf16_t*)take((size_t)T * Bb * Dd * 2);
    float*  xg      = (float*) take((size_t)T * Bb * 4 * Dd * 4);
    bf16_t* spk0    = (bf16_t*)take((size_t)T * Bb * Dd * 2);  // time-reversed
    bf16_t* spk1    = (bf16_t*)take((size_t)T * Bb * Dd * 2);
    float*  mem_f   = (float*) take((size_t)Bb * Dd * 4);
    float*  syn_f   = (float*) take((size_t)Bb * Dd * 4);
    bf16_t* membf[2];
    membf[0] = (bf16_t*)take((size_t)Bb * Dd * 2);
    membf[1] = (bf16_t*)take((size_t)Bb * Dd * 2);

    auto conv = [&](const float* src, bf16_t* dst, size_t n) {
        f32_to_bf16_kernel<<<dim3((unsigned)((n + 255) / 256)), 256, 0, stream>>>(src, dst, n);
    };
    for (int l = 0; l < 2; ++l) {
        conv(lin_w[l], lin_bf[l], (size_t)Dd * Dd);
        conv(wih[l],   wih_bf[l], (size_t)4 * Dd * Dd);
        conv(whh[l],   whh_bf[l], (size_t)4 * Dd * Dd);
    }
    conv(dec_w, decw_bf, (size_t)Vv * Dd);

    embed_spike_kernel<<<T * Bb, 256, 0, stream>>>(data, emb, x_spk, Dd);

    const int M = T * Bb;  // 2048
    for (int l = 0; l < 2; ++l) {
        const bf16_t* a_in = (l == 0) ? x_spk : spk0;
        bf16_t* spk_out_base = (l == 0) ? spk0 : spk1;
        gemm_lds_kernel<true><<<dim3(M / 128, Dd / 64), 128, 0, stream>>>(
            a_in, lin_bf[l], lin_b[l], nullptr, y_bf, M, Dd, Dd);
        gemm_lds_kernel<false><<<dim3(M / 128, 4 * Dd / 64), 128, 0, stream>>>(
            y_bf, wih_bf[l], bih[l], bhh[l], xg, M, 4 * Dd, Dd);
        hipMemsetAsync(mem_f,    0, (size_t)Bb * Dd * 4, stream);
        hipMemsetAsync(syn_f,    0, (size_t)Bb * Dd * 4, stream);
        hipMemsetAsync(membf[0], 0, (size_t)Bb * Dd * 2, stream);
        for (int t = 0; t < T; ++t) {
            int outT = (l == 0) ? (T - 1 - t) : t;  // layer0 stored reversed
            slstm_step_kernel<<<Dd / 16, 64, 0, stream>>>(
                xg + (size_t)t * Bb * 4 * Dd, whh_bf[l],
                membf[t & 1], membf[(t & 1) ^ 1],
                mem_f, syn_f,
                spk_out_base + (size_t)outT * Bb * Dd,
                thr[l], Dd);
        }
    }

    decoder_kernel<<<dim3(Bb / 2, Vv / 64), 128, 0, stream>>>(
        spk1, decw_bf, dec_b, out, beta, Bb, Dd, Vv, T);
}
